// RelationAwareAttention_11338713661449
// MI455X (gfx1250) — matrix operand order repaired and checked by reading, not compiled, not run on hardware
//
#include <hip/hip_runtime.h>
#include <hip/hip_bf16.h>

// B=8, L=1024, D=1024, H=16, HD=64, R=8
typedef __attribute__((ext_vector_type(16))) __bf16 v16bf;
typedef __attribute__((ext_vector_type(8)))  float  v8f;

struct Frag32B { uint4 lo, hi; };

__device__ __forceinline__ unsigned short f2bf(float f) {
  unsigned int u = __builtin_bit_cast(unsigned int, f);
  u += 0x7fffu + ((u >> 16) & 1u);   // round-to-nearest-even
  return (unsigned short)(u >> 16);
}

__device__ __forceinline__ v16bf load_frag(const unsigned short* p0,
                                           const unsigned short* p1) {
  Frag32B u;
  u.lo = *(const uint4*)p0;
  u.hi = *(const uint4*)p1;
  return __builtin_bit_cast(v16bf, u);
}

__device__ __forceinline__ v8f wmma_bf16(v16bf a, v16bf b, v8f c) {
  return __builtin_amdgcn_wmma_f32_16x16x32_bf16(false, a, false, b,
                                                 (short)0, c, false, false);
}

// Async copy: 16B per lane, global -> LDS, tracked by ASYNCcnt (no VGPR data).
__device__ __forceinline__ void async_b128(unsigned lds_off, const void* gsrc) {
  asm volatile("global_load_async_to_lds_b128 %0, %1, off"
               :: "v"(lds_off), "v"(gsrc) : "memory");
}
__device__ __forceinline__ void wait_async0() {
#if __has_builtin(__builtin_amdgcn_s_wait_asynccnt)
  __builtin_amdgcn_s_wait_asynccnt(0);
#else
  asm volatile("s_wait_asynccnt 0" ::: "memory");
#endif
}
__device__ __forceinline__ unsigned lds_addr(const void* p) {
  return (unsigned)(unsigned long long)p;  // low 32 bits = wave LDS offset
}

__device__ __forceinline__ float redmax16(float v) {
  v = fmaxf(v, __shfl_xor(v, 1, 32));
  v = fmaxf(v, __shfl_xor(v, 2, 32));
  v = fmaxf(v, __shfl_xor(v, 4, 32));
  v = fmaxf(v, __shfl_xor(v, 8, 32));
  return v;
}
__device__ __forceinline__ float redsum16(float v) {
  v += __shfl_xor(v, 1, 32);
  v += __shfl_xor(v, 2, 32);
  v += __shfl_xor(v, 4, 32);
  v += __shfl_xor(v, 8, 32);
  return v;
}

// ---------------------------------------------------------------------------
// Prep: x -> bf16; Wq/Wk/Wv/Wo -> transposed bf16 (WT[n,k] = W[k,n])
// ---------------------------------------------------------------------------
__global__ void __launch_bounds__(256) prep_kernel(
    const float* __restrict__ x,
    const float* __restrict__ Wq, const float* __restrict__ Wk,
    const float* __restrict__ Wv, const float* __restrict__ Wo,
    unsigned short* __restrict__ xb,
    unsigned short* __restrict__ WqT, unsigned short* __restrict__ WkT,
    unsigned short* __restrict__ WvT, unsigned short* __restrict__ WoT) {
  const int tid = blockIdx.x * blockDim.x + threadIdx.x;
  const int nt  = gridDim.x * blockDim.x;
  for (int i = tid; i < 8192 * 1024; i += nt) xb[i] = f2bf(x[i]);
  for (int i = tid; i < 1024 * 1024; i += nt) {
    int n = i >> 10, k = i & 1023;
    int src = k * 1024 + n;
    WqT[i] = f2bf(Wq[src]);
    WkT[i] = f2bf(Wk[src]);
    WvT[i] = f2bf(Wv[src]);
    WoT[i] = f2bf(Wo[src]);
  }
}

// ---------------------------------------------------------------------------
// Tiled bf16 WMMA GEMM with async-to-LDS double buffering.
//   MODE 0: q layout [B,H,L,HD], scaled by 1/sqrt(HD)
//   MODE 1: k layout [B,H,L,HD]
//   MODE 2: v transposed layout [B,H,HD,L]
//   MODE 3: fp32 row-major output [B*L, D]
// Block: 128 threads (4 waves); tile 64(M) x 128(N); K staged 32 at a time.
// ---------------------------------------------------------------------------
template <int MODE>
__global__ void __launch_bounds__(128) gemm_kernel(
    const unsigned short* __restrict__ A,
    const unsigned short* __restrict__ BT,
    const float* __restrict__ bias,
    void* __restrict__ outp) {
  __shared__ alignas(16) unsigned short At[2][64 * 32];
  __shared__ alignas(16) unsigned short Bt[2][128 * 32];
  const int t = threadIdx.x;
  const int w = t >> 5, lane = t & 31, lo = lane & 15, hi = lane >> 4;
  const int m0 = blockIdx.y * 64;
  const int n0 = blockIdx.x * 128;

  v8f zero = {};
  v8f acc[4][2];
  for (int i = 0; i < 4; ++i)
    for (int j = 0; j < 2; ++j) acc[i][j] = zero;

  auto issue_tile = [&](int buf, int k0) {
#pragma unroll
    for (int i = 0; i < 2; ++i) {
      int u = t + i * 128, row = u >> 2, seg = u & 3;   // A: 64 rows x 4 chunks
      async_b128(lds_addr(&At[buf][row * 32 + seg * 8]),
                 A + (m0 + row) * 1024 + k0 + seg * 8);
    }
#pragma unroll
    for (int i = 0; i < 4; ++i) {
      int u = t + i * 128, row = u >> 2, seg = u & 3;   // B: 128 rows x 4 chunks
      async_b128(lds_addr(&Bt[buf][row * 32 + seg * 8]),
                 BT + (n0 + row) * 1024 + k0 + seg * 8);
    }
  };

  issue_tile(0, 0);
  for (int it = 0; it < 32; ++it) {
    wait_async0();
    __syncthreads();                       // tile `it` resident, prev reads done
    if (it + 1 < 32) issue_tile((it + 1) & 1, (it + 1) * 32);
    if (it + 2 < 32) {                     // distance-2 L2 prefetch
      __builtin_prefetch(A + (m0 + (t >> 1)) * 1024 + (it + 2) * 32, 0, 0);
      __builtin_prefetch(BT + (n0 + t) * 1024 + (it + 2) * 32, 0, 0);
    }
    const unsigned short* Ac = At[it & 1];
    const unsigned short* Bc = Bt[it & 1];

    v16bf bf[2];
#pragma unroll
    for (int j = 0; j < 2; ++j) {
      const unsigned short* p = Bc + (w * 32 + j * 16 + lo) * 32 + hi * 16;
      bf[j] = load_frag(p, p + 8);
    }
#pragma unroll
    for (int i = 0; i < 4; ++i) {
      const unsigned short* p = Ac + (i * 16 + lo) * 32 + hi * 16;
      v16bf af = load_frag(p, p + 8);
#pragma unroll
      for (int j = 0; j < 2; ++j) acc[i][j] = wmma_bf16(af, bf[j], acc[i][j]);
    }
  }

  // Epilogue: bias, optional scale, scatter per mode.
  const float scale = (MODE == 0) ? 0.125f : 1.0f;
#pragma unroll
  for (int j = 0; j < 2; ++j) {
    const int n = n0 + w * 32 + j * 16 + lo;
    const float bv = bias[n];
    const int h = (n >> 6) & 15, d = n & 63;
#pragma unroll
    for (int i = 0; i < 4; ++i) {
#pragma unroll
      for (int r = 0; r < 8; ++r) {
        const int m = m0 + i * 16 + r + hi * 8;
        const float val = acc[i][j][r] + bv;
        if (MODE == 3) {
          ((float*)outp)[m * 1024 + n] = val;
        } else {
          const int b = m >> 10, l = m & 1023;
          unsigned short* o = (unsigned short*)outp;
          if (MODE == 2)
            o[((b * 16 + h) * 64 + d) * 1024 + l] = f2bf(val);
          else
            o[((b * 16 + h) * 1024 + l) * 64 + d] = f2bf(val * scale);
        }
      }
    }
  }
}

// ---------------------------------------------------------------------------
// Flash attention: ctx = softmax(q k^T) v  (q pre-scaled by 1/8).
// Grid (L/64, H, B); block 128 (4 waves, 16 q-rows each); key tiles of 64,
// double-buffered via async-to-LDS.
// ---------------------------------------------------------------------------
__global__ void __launch_bounds__(128) flash_kernel(
    const unsigned short* __restrict__ qb,
    const unsigned short* __restrict__ kb,
    const unsigned short* __restrict__ vtb,
    unsigned short* __restrict__ ctxb) {
  __shared__ alignas(16) unsigned short Kt[2][64 * 64];    // [key][hd]
  __shared__ alignas(16) unsigned short Vt[2][64 * 64];    // [hd][key]
  __shared__ alignas(16) unsigned short Pt[4][16 * 64];    // per-wave P tile
  const int t = threadIdx.x;
  const int w = t >> 5, lane = t & 31, lo = lane & 15, hi = lane >> 4;
  const int bh = blockIdx.z * 16 + blockIdx.y;
  const int q0 = blockIdx.x * 64;

  // Q fragments (A layout) held in registers for the whole kernel.
  v16bf qf[2];
  {
    const unsigned short* qrow = qb + (bh * 1024 + q0 + w * 16 + lo) * 64;
#pragma unroll
    for (int c = 0; c < 2; ++c) {
      const unsigned short* p = qrow + c * 32 + hi * 16;
      qf[c] = load_frag(p, p + 8);
    }
  }

  auto issue_kv = [&](int buf, int kj) {
#pragma unroll
    for (int i = 0; i < 4; ++i) {
      int u = t + i * 128, row = u >> 3, seg = u & 7;  // 64 rows x 8 chunks
      async_b128(lds_addr(&Kt[buf][row * 64 + seg * 8]),
                 kb + (bh * 1024 + kj + row) * 64 + seg * 8);
      async_b128(lds_addr(&Vt[buf][row * 64 + seg * 8]),
                 vtb + (bh * 64 + row) * 1024 + kj + seg * 8);
    }
  };

  v8f zero = {};
  v8f acc[4];
  for (int f = 0; f < 4; ++f) acc[f] = zero;
  float mrow[8], srow[8];
#pragma unroll
  for (int r = 0; r < 8; ++r) { mrow[r] = -1e30f; srow[r] = 0.0f; }

  issue_kv(0, 0);
  for (int it = 0; it < 16; ++it) {
    wait_async0();
    __syncthreads();                     // tile `it` resident, prev reads done
    if (it + 1 < 16) issue_kv((it + 1) & 1, (it + 1) * 64);
    const unsigned short* Kc = Kt[it & 1];
    const unsigned short* Vc = Vt[it & 1];

    // S = Q K^T for 4 key subtiles of 16.
    v8f sf[4];
#pragma unroll
    for (int sub = 0; sub < 4; ++sub) {
      v8f s = zero;
#pragma unroll
      for (int c = 0; c < 2; ++c) {
        const unsigned short* p = Kc + (sub * 16 + lo) * 64 + c * 32 + hi * 16;
        v16bf kf = load_frag(p, p + 8);
        s = wmma_bf16(qf[c], kf, s);
      }
      sf[sub] = s;
    }

    // Online softmax row stats (rows live in 16-lane groups under wave32).
    float corr[8];
#pragma unroll
    for (int r = 0; r < 8; ++r) {
      float tm = fmaxf(fmaxf(sf[0][r], sf[1][r]), fmaxf(sf[2][r], sf[3][r]));
      tm = redmax16(tm);
      float nm = fmaxf(mrow[r], tm);
      corr[r] = __expf(mrow[r] - nm);
      mrow[r] = nm;
    }
#pragma unroll
    for (int r = 0; r < 8; ++r) {
      float ps = 0.0f;
#pragma unroll
      for (int sub = 0; sub < 4; ++sub) {
        float p = __expf(sf[sub][r] - mrow[r]);
        Pt[w][(r + hi * 8) * 64 + sub * 16 + lo] = f2bf(p);  // C -> A relayout
        ps += p;
      }
      ps = redsum16(ps);
      srow[r] = srow[r] * corr[r] + ps;
    }
#pragma unroll
    for (int f = 0; f < 4; ++f)
#pragma unroll
      for (int r = 0; r < 8; ++r) acc[f][r] *= corr[r];

    // ctx += P @ V (A frags from own-wave Pt; same-wave LDS is in-order).
#pragma unroll
    for (int c = 0; c < 2; ++c) {
      const unsigned short* pa = &Pt[w][lo * 64 + c * 32 + hi * 16];
      v16bf af = load_frag(pa, pa + 8);
#pragma unroll
      for (int f = 0; f < 4; ++f) {
        const unsigned short* pb = Vc + (f * 16 + lo) * 64 + c * 32 + hi * 16;
        v16bf vf = load_frag(pb, pb + 8);
        acc[f] = wmma_bf16(af, vf, acc[f]);
      }
    }
  }

  const int b = bh >> 4, h = bh & 15;
#pragma unroll
  for (int r = 0; r < 8; ++r) {
    const float inv = 1.0f / srow[r];
    const int l = q0 + w * 16 + r + hi * 8;
#pragma unroll
    for (int f = 0; f < 4; ++f)
      ctxb[(b * 1024 + l) * 1024 + h * 64 + f * 16 + lo] =
          f2bf(acc[f][r] * inv);
  }
}

// ---------------------------------------------------------------------------
// relation_scores[b,r] = (1/L) * sum_h Wrel[h,r] + brel[r]
// (softmax rows sum to 1 => attn.mean over (L,L) == 1/L exactly)
// ---------------------------------------------------------------------------
__global__ void rel_kernel(const float* __restrict__ Wrel,
                           const float* __restrict__ brel,
                           float* __restrict__ out) {
  const int t = threadIdx.x;      // 64 threads: b = t>>3, r = t&7
  const int r = t & 7;
  float s = 0.0f;
#pragma unroll
  for (int h = 0; h < 16; ++h) s += Wrel[h * 8 + r];
  out[8 * 1024 * 1024 + t] = s * (1.0f / 1024.0f) + brel[r];
}

extern "C" void kernel_launch(void* const* d_in, const int* in_sizes, int n_in,
                              void* d_out, int out_size, void* d_ws, size_t ws_size,
                              hipStream_t stream) {
  (void)in_sizes; (void)n_in; (void)out_size; (void)ws_size;
  const float* x    = (const float*)d_in[0];
  const float* Wq   = (const float*)d_in[1];
  const float* bq   = (const float*)d_in[2];
  const float* Wk   = (const float*)d_in[3];
  const float* bk   = (const float*)d_in[4];
  const float* Wv   = (const float*)d_in[5];
  const float* bv   = (const float*)d_in[6];
  const float* Wo   = (const float*)d_in[7];
  const float* bo   = (const float*)d_in[8];
  const float* Wrel = (const float*)d_in[9];
  const float* brel = (const float*)d_in[10];
  float* out = (float*)d_out;

  char* ws = (char*)d_ws;
  unsigned short* xb   = (unsigned short*)ws;                       // 16 MB
  unsigned short* WqT  = (unsigned short*)(ws + (16u << 20));       // 2 MB
  unsigned short* WkT  = WqT + 1024 * 1024;
  unsigned short* WvT  = WkT + 1024 * 1024;
  unsigned short* WoT  = WvT + 1024 * 1024;
  unsigned short* qbuf = WoT + 1024 * 1024;                         // 16 MB
  unsigned short* kbuf = qbuf + 8 * 1024 * 1024;                    // 16 MB
  unsigned short* vtb  = kbuf + 8 * 1024 * 1024;                    // 16 MB
  unsigned short* ctxb = vtb + 8 * 1024 * 1024;                     // 16 MB

  prep_kernel<<<1024, 256, 0, stream>>>(x, Wq, Wk, Wv, Wo,
                                        xb, WqT, WkT, WvT, WoT);
  dim3 g(8, 128);  // N/128, M/64
  gemm_kernel<0><<<g, 128, 0, stream>>>(xb, WqT, bq, qbuf);
  gemm_kernel<1><<<g, 128, 0, stream>>>(xb, WkT, bk, kbuf);
  gemm_kernel<2><<<g, 128, 0, stream>>>(xb, WvT, bv, vtb);
  flash_kernel<<<dim3(16, 16, 8), 128, 0, stream>>>(qbuf, kbuf, vtb, ctxb);
  gemm_kernel<3><<<g, 128, 0, stream>>>(ctxb, WoT, bo, out);
  rel_kernel<<<1, 64, 0, stream>>>(Wrel, brel, out);
}
